// seg_model_3264175145576
// MI455X (gfx1250) — matrix operand order repaired and checked by reading, not compiled, not run on hardware
//
#include <hip/hip_runtime.h>

typedef __attribute__((ext_vector_type(16))) _Float16 v16h;
typedef __attribute__((ext_vector_type(8)))  _Float16 v8h;
typedef __attribute__((ext_vector_type(8)))  float    v8f;

// ---------------------------------------------------------------------------
// Batched GEMM: C[b] = epilogue( A[b] (MxK) * B[b] (KxN) )
// INVARIANTS (host-guaranteed via padded buffers): M%128==0, N%64==0, K%32==0.
// -> staging has NO guards / NO slow paths; the only guard is gnc < Nstore at
// the final store (Nstore <= N handles the 50-wide output layer).
// fp32 in global memory, f16 staged in LDS (all b128), f32 WMMA accumulate.
// Block: 256 threads (8 waves), 128x64 C tile, wave tile 32x32 (2x2 WMMA).
// epi: 0 -> C = alpha*acc
//      1 -> C = exp(-(yy[m] - 2*acc + yy[n]))   (gram / adjacency)
//      2 -> C = acc + bias[n], optional ReLU    (dense layer)
//      3 -> C = alpha*acc + beta*D
// ---------------------------------------------------------------------------
template <int TRANSB>
__global__ __launch_bounds__(256) void gemm_wmma_f16(
    const float* __restrict__ A, const float* __restrict__ B,
    float* __restrict__ C,
    int M, int N, int K, int Nstore,
    long sA, long sB, long sC,
    int lda, int ldb, int ldc,
    float alpha, float beta,
    const float* __restrict__ D, long sD, int ldd,
    const float* __restrict__ bias,
    const float* __restrict__ yy,
    int epi, int relu)
{
    // Row stride 48 halves = 96 B (multiple of 16B/32B) -> aligned b128 access.
    __shared__ __align__(32) _Float16 As[128][48];  // [m][k]
    __shared__ __align__(32) _Float16 Bs[64][48];   // [n][k]  (k-contig per col)

    const int b = blockIdx.z;
    A += (long)b * sA;
    B += (long)b * sB;
    C += (long)b * sC;
    const float* Db = D + (long)b * sD;
    if (yy) yy += (long)b * M;      // gram is square: yy length M==N per batch

    const int tid  = threadIdx.x;
    const int wave = tid >> 5;
    const int lane = tid & 31;
    const int wm = (wave & 3) * 32;   // wave row offset within 128
    const int wn = (wave >> 2) * 32;  // wave col offset within 64
    const int m0 = blockIdx.y * 128;
    const int n0 = blockIdx.x * 64;

    v8f acc00 = {}; v8f acc01 = {};
    v8f acc10 = {}; v8f acc11 = {};

    const int row = lane & 15;
    const int ha  = (lane < 16) ? 0 : 16;   // A halves: K 0-15 vs 16-31
    const int kb  = (lane < 16) ? 0 : 16;   // B halves: K 0-15 vs 16-31

    for (int k0 = 0; k0 < K; k0 += 32) {
        // ---- stage A tile: 128 rows x 32 k; unguarded, 2x b128 store --------
        #pragma unroll
        for (int it = 0; it < 2; ++it) {
            const int r  = (tid >> 2) + it * 64;
            const int kk = (tid & 3) * 8;
            const float* ap = A + (long)(m0 + r) * lda + k0 + kk;
            float f[8];
            #pragma unroll
            for (int j = 0; j < 8; ++j) f[j] = ap[j];
            if (k0 + 32 < K) __builtin_prefetch(ap + 32, 0, 1);
            v8h h;
            #pragma unroll
            for (int j = 0; j < 8; ++j) h[j] = (_Float16)f[j];
            *(v8h*)&As[r][kk] = h;
        }

        // ---- stage B tile into n-major LDS: 64 cols x 32 k; unguarded -------
        {
            float f[8];
            int c, kk;
            if (TRANSB) {
                // B in memory: [N rows x K cols]; k contiguous per row.
                c  = tid >> 2;
                kk = (tid & 3) * 8;
                const float* bp = B + (long)(n0 + c) * ldb + k0 + kk;
                #pragma unroll
                for (int j = 0; j < 8; ++j) f[j] = bp[j];
                if (k0 + 32 < K) __builtin_prefetch(bp + 32, 0, 1);
            } else {
                // B in memory: [K rows x N cols]; lanes cover consecutive n
                // at the same k -> coalesced.
                c  = tid & 63;
                kk = (tid >> 6) * 8;
                const float* bp = B + (long)(k0 + kk) * ldb + n0 + c;
                #pragma unroll
                for (int j = 0; j < 8; ++j) f[j] = bp[(long)j * ldb];
            }
            v8h h;
            #pragma unroll
            for (int j = 0; j < 8; ++j) h[j] = (_Float16)f[j];
            *(v8h*)&Bs[c][kk] = h;
        }
        __syncthreads();

        // ---- fragments: all b128 LDS loads ---------------------------------
        v8h a00 = *(const v8h*)&As[wm + row][ha];
        v8h a01 = *(const v8h*)&As[wm + row][8 + ha];
        v8h a10 = *(const v8h*)&As[wm + 16 + row][ha];
        v8h a11 = *(const v8h*)&As[wm + 16 + row][8 + ha];
        v16h af0 = __builtin_shufflevector(a00, a01,
            0,1,2,3,4,5,6,7,8,9,10,11,12,13,14,15);
        v16h af1 = __builtin_shufflevector(a10, a11,
            0,1,2,3,4,5,6,7,8,9,10,11,12,13,14,15);
        v16h bf0 = *(const v16h*)&Bs[wn + row][kb];
        v16h bf1 = *(const v16h*)&Bs[wn + 16 + row][kb];

        acc00 = __builtin_amdgcn_wmma_f32_16x16x32_f16(
            false, af0, false, bf0, (short)0, acc00, false, false);
        acc01 = __builtin_amdgcn_wmma_f32_16x16x32_f16(
            false, af0, false, bf1, (short)0, acc01, false, false);
        acc10 = __builtin_amdgcn_wmma_f32_16x16x32_f16(
            false, af1, false, bf0, (short)0, acc10, false, false);
        acc11 = __builtin_amdgcn_wmma_f32_16x16x32_f16(
            false, af1, false, bf1, (short)0, acc11, false, false);
        __syncthreads();
    }

    // ---- epilogue + store (C/D layout: VGPR r -> M = r or r+8) -------------
    const int rofs = (lane < 16) ? 0 : 8;
    #pragma unroll
    for (int ms = 0; ms < 2; ++ms) {
        #pragma unroll
        for (int ns = 0; ns < 2; ++ns) {
            v8f a = (ms == 0) ? (ns == 0 ? acc00 : acc01)
                              : (ns == 0 ? acc10 : acc11);
            const int gnc = n0 + wn + ns * 16 + row;
            if (gnc >= Nstore) continue;    // only guard (padded output cols)
            #pragma unroll
            for (int r = 0; r < 8; ++r) {
                const int gm = m0 + wm + ms * 16 + rofs + r;
                float v = a[r];
                float o;
                if (epi == 1) {
                    float d2 = yy[gm] - 2.0f * v + yy[gnc];
                    o = expf(-d2);
                } else if (epi == 2) {
                    o = v + bias[gnc];
                    if (relu) o = fmaxf(o, 0.0f);
                } else if (epi == 3) {
                    o = alpha * v + beta * Db[(long)gm * ldd + gnc];
                } else {
                    o = alpha * v;
                }
                C[(long)gm * ldc + gnc] = o;
            }
        }
    }
}

// ---------------------------------------------------------------------------
// Support kernels
// ---------------------------------------------------------------------------
// dst is nslices groups of finP rows (each fout wide); first finR rows of each
// group copied from src (slices of finR rows), rest zero-filled.
__global__ void pad_slices_kernel(const float* __restrict__ src,
                                  float* __restrict__ dst,
                                  long total, int finR, int finP, int fout)
{
    long i = (long)blockIdx.x * blockDim.x + threadIdx.x;
    if (i >= total) return;
    int n = (int)(i % fout);
    long t = i / fout;
    int f = (int)(t % finP);
    long s = t / finP;
    dst[i] = (f < finR) ? src[(s * finR + f) * (long)fout + n] : 0.0f;
}

__global__ void rowsumsq_kernel(const float* __restrict__ y,
                                float* __restrict__ yy, long rows, int F)
{
    long i = (long)blockIdx.x * blockDim.x + threadIdx.x;
    if (i >= rows) return;
    const float* p = y + i * (long)F;
    float s = 0.0f;
    for (int f = 0; f < F; ++f) { float v = p[f]; s += v * v; }
    yy[i] = s;
}

__global__ __launch_bounds__(256) void rowsum_dinv_kernel(
    const float* __restrict__ A, float* __restrict__ dinv, int N)
{
    long rowBase = (long)blockIdx.x * N;
    __shared__ float red[256];
    float s = 0.0f;
    for (int m = threadIdx.x; m < N; m += 256) s += A[rowBase + m];
    red[threadIdx.x] = s;
    __syncthreads();
    for (int w = 128; w > 0; w >>= 1) {
        if ((int)threadIdx.x < w) red[threadIdx.x] += red[threadIdx.x + w];
        __syncthreads();
    }
    if (threadIdx.x == 0) dinv[blockIdx.x] = rsqrtf(red[0] + 1e-12f);
}

__global__ void laplacian_kernel(float* __restrict__ A,
                                 const float* __restrict__ dinv,
                                 int N, long total)
{
    long i = (long)blockIdx.x * blockDim.x + threadIdx.x;
    if (i >= total) return;
    long nn = (long)N * N;
    long bb = i / nn;
    long nm = i - bb * nn;
    int n = (int)(nm / N);
    int m = (int)(nm - (long)n * N);
    float v = -dinv[bb * N + n] * A[i] * dinv[bb * N + m];
    if (n == m) v += 1.0f;
    A[i] = v;
}

__global__ void copy_cols_kernel(const float* __restrict__ src,
                                 float* __restrict__ dst,
                                 long total, int F, int lds, int ldd,
                                 int rowsPerBatch, long sSrc, long sDst)
{
    long i = (long)blockIdx.x * blockDim.x + threadIdx.x;
    if (i >= total) return;
    int f = (int)(i % F);
    long r = i / F;
    int n = (int)(r % rowsPerBatch);
    int b = (int)(r / rowsPerBatch);
    dst[(long)b * sDst + (long)n * ldd + f] =
        src[(long)b * sSrc + (long)n * lds + f];
}

__global__ void zero_kernel(float* p, int n)
{
    int i = blockIdx.x * blockDim.x + threadIdx.x;
    if (i < n) p[i] = 0.0f;
}

// r[b,f,g] = sum_n out[b,n,f] * t[b,n,g]; atomically accumulate r^2
__global__ __launch_bounds__(256) void reg_reduce_kernel(
    const float* __restrict__ out, const float* __restrict__ t,
    float* __restrict__ reg, int N, int F, int G, int ldT,
    long sOut, long sT)
{
    int fg = blockIdx.x;
    int b  = blockIdx.y;
    int f = fg / G, g = fg % G;
    const float* ob = out + (long)b * sOut;
    const float* tb = t   + (long)b * sT;
    __shared__ float red[256];
    float s = 0.0f;
    for (int n = threadIdx.x; n < N; n += 256)
        s += ob[(long)n * F + f] * tb[(long)n * ldT + g];
    red[threadIdx.x] = s;
    __syncthreads();
    for (int w = 128; w > 0; w >>= 1) {
        if ((int)threadIdx.x < w) red[threadIdx.x] += red[threadIdx.x + w];
        __syncthreads();
    }
    if (threadIdx.x == 0) { float r = red[0]; atomicAdd(reg, r * r); }
}

// ---------------------------------------------------------------------------
// Host orchestration
// ---------------------------------------------------------------------------
static inline void launch_gemm(hipStream_t s,
    const float* A, const float* B, float* C,
    int M, int N, int K, int Nstore, long sA, long sB, long sC,
    int lda, int ldb, int ldc, int transB,
    float alpha, float beta, const float* D, long sD, int ldd,
    const float* bias, const float* yy, int epi, int relu, int batches)
{
    dim3 g((N + 63) / 64, (M + 127) / 128, batches), blk(256);
    if (transB)
        gemm_wmma_f16<1><<<g, blk, 0, s>>>(A, B, C, M, N, K, Nstore,
                                           sA, sB, sC, lda, ldb, ldc,
                                           alpha, beta, D, sD, ldd,
                                           bias, yy, epi, relu);
    else
        gemm_wmma_f16<0><<<g, blk, 0, s>>>(A, B, C, M, N, K, Nstore,
                                           sA, sB, sC, lda, ldb, ldc,
                                           alpha, beta, D, sD, ldd,
                                           bias, yy, epi, relu);
}

struct LayerCtx {
    hipStream_t s;
    int Bb, Nn;
    float *Abuf, *dinv, *yy, *xk, *tbuf, *regs;
    const float* x0;   // padded [B,Nn,64], cols >=22 are zero
};

// Fin: padded feature width (lda of src; multiple of 32, and of 64 when it is
// the N-dim of a chain GEMM).  FoutPad: padded dense output width (mult 64).
// FoutStore: real stored width (ldc of dst).
static void run_layer(const LayerCtx& p, const float* src, int Fin, int K,
                      const float* W, const float* bias,
                      float* dst, int FoutPad, int FoutStore, int relu, int li)
{
    const int Bb = p.Bb, Nn = p.Nn;
    const long rows = (long)Bb * Nn;

    // yy[b,n] = ||src_row||^2  (padded cols are zero -> exact)
    rowsumsq_kernel<<<(rows + 255) / 256, 256, 0, p.s>>>(src, p.yy, rows, Fin);

    // A = exp(-d2) via gram GEMM (NT) with fused epilogue
    launch_gemm(p.s, src, src, p.Abuf, Nn, Nn, Fin, Nn,
                (long)Nn * Fin, (long)Nn * Fin, (long)Nn * Nn,
                Fin, Fin, Nn, /*transB=*/1, 1.f, 0.f,
                p.Abuf, 0, 1, nullptr, p.yy, /*epi=*/1, 0, Bb);

    // deg -> dinv
    rowsum_dinv_kernel<<<(unsigned)rows, 256, 0, p.s>>>(p.Abuf, p.dinv, Nn);

    // L = I - dinv A dinv  (in place over Abuf)
    long tot = (long)Bb * Nn * Nn;
    laplacian_kernel<<<(tot + 255) / 256, 256, 0, p.s>>>(p.Abuf, p.dinv, Nn, tot);

    // Chebyshev expansion into xk (column-sliced [B,Nn,K*Fin])
    const float* dense_src;
    int ldA;
    const int Kdim = K * Fin;
    if (K == 1) {
        dense_src = src; ldA = Fin;
    } else {
        const int ldxk = K * Fin;
        long totc = rows * (long)Fin;
        copy_cols_kernel<<<(totc + 255) / 256, 256, 0, p.s>>>(
            src, p.xk, totc, Fin, Fin, ldxk, Nn,
            (long)Nn * Fin, (long)Nn * ldxk);
        // x1 = L @ x0
        launch_gemm(p.s, p.Abuf, src, p.xk + Fin, Nn, Fin, Nn, Fin,
                    (long)Nn * Nn, (long)Nn * Fin, (long)Nn * ldxk,
                    Nn, Fin, ldxk, 0, 1.f, 0.f,
                    p.xk, 0, 1, nullptr, nullptr, 0, 0, Bb);
        // x_k = 2 L x_{k-1} - x_{k-2}
        for (int k = 2; k < K; ++k) {
            launch_gemm(p.s, p.Abuf, p.xk + (long)(k - 1) * Fin,
                        p.xk + (long)k * Fin, Nn, Fin, Nn, Fin,
                        (long)Nn * Nn, (long)Nn * ldxk, (long)Nn * ldxk,
                        Nn, ldxk, ldxk, 0, 2.f, -1.f,
                        p.xk + (long)(k - 2) * Fin, (long)Nn * ldxk, ldxk,
                        nullptr, nullptr, 3, 0, Bb);
        }
        dense_src = p.xk; ldA = ldxk;
    }

    // dense: dst = relu(xk @ W + b); store only FoutStore columns
    launch_gemm(p.s, dense_src, W, dst, Nn, FoutPad, Kdim, FoutStore,
                (long)Nn * ldA, 0, (long)Nn * FoutStore,
                ldA, FoutPad, FoutStore, 0, 1.f, 0.f,
                dst, 0, 1, bias, nullptr, /*epi=*/2, relu, Bb);

    // reg: t = L @ x0pad (cols >=22 stay zero) then sum((out^T t)^2)
    launch_gemm(p.s, p.Abuf, p.x0, p.tbuf, Nn, 64, Nn, 64,
                (long)Nn * Nn, (long)Nn * 64, (long)Nn * 64,
                Nn, 64, 64, 0, 1.f, 0.f,
                p.tbuf, 0, 1, nullptr, nullptr, 0, 0, Bb);
    dim3 rg(FoutStore * 22, Bb);
    reg_reduce_kernel<<<rg, 256, 0, p.s>>>(dst, p.tbuf, p.regs + li,
                                           Nn, FoutStore, 22, 64,
                                           (long)Nn * FoutStore, (long)Nn * 64);
}

extern "C" void kernel_launch(void* const* d_in, const int* in_sizes, int n_in,
                              void* d_out, int out_size, void* d_ws, size_t ws_size,
                              hipStream_t stream)
{
    (void)in_sizes; (void)n_in; (void)out_size; (void)ws_size;
    const int Bb = 8, Nn = 2048;
    const float* x = (const float*)d_in[0];
    const float* W[6];
    const float* bs[6];
    for (int i = 0; i < 6; ++i) {
        W[i]  = (const float*)d_in[1 + 2 * i];
        bs[i] = (const float*)d_in[2 + 2 * i];
    }
    float* out  = (float*)d_out;
    float* regs = out + (long)Bb * Nn * 50;

    // workspace carve (floats)
    float* ws = (float*)d_ws;
    long off = 0;
    float* Abuf  = ws + off; off += (long)Bb * Nn * Nn;    // A then L (in place)
    float* dinv  = ws + off; off += (long)Bb * Nn;
    float* yy    = ws + off; off += (long)Bb * Nn;
    float* xk    = ws + off; off += (long)Bb * Nn * 1536;  // cheb concat (max)
    float* buf0  = ws + off; off += (long)Bb * Nn * 1152;
    float* buf1  = ws + off; off += (long)Bb * Nn * 1152;
    float* x1b   = ws + off; off += (long)Bb * Nn * 128;   // layer-1 out (skip)
    float* tbuf  = ws + off; off += (long)Bb * Nn * 64;    // L @ x0pad for reg
    float* x0pad = ws + off; off += (long)Bb * Nn * 64;    // x padded 22->64
    float* w1pad = ws + off; off += (long)6 * 64 * 128;    // W1 rows 22->64/slice
    float* w6pad = ws + off; off += (long)128 * 64;        // W6 cols 50->64

    zero_kernel<<<1, 32, 0, stream>>>(regs, 6);

    const long rows = (long)Bb * Nn;
    // Padding (all GEMM dims become multiples of 128/64/32):
    long tp = rows * 64;
    pad_slices_kernel<<<(tp + 255) / 256, 256, 0, stream>>>(
        x, x0pad, tp, 22, 64, 1);
    tp = (long)6 * 64 * 128;
    pad_slices_kernel<<<(tp + 255) / 256, 256, 0, stream>>>(
        W[0], w1pad, tp, 22, 64, 128);
    tp = (long)128 * 64;
    pad_slices_kernel<<<(tp + 255) / 256, 256, 0, stream>>>(
        W[5], w6pad, tp, 50, 64, 1);

    LayerCtx p{stream, Bb, Nn, Abuf, dinv, yy, xk, tbuf, regs, x0pad};

    run_layer(p, x0pad, 64,  6, w1pad, bs[0], x1b,  128, 128, 1, 0);
    run_layer(p, x1b,   128, 5, W[1],  bs[1], buf0, 512, 512, 1, 1);
    run_layer(p, buf0,  512, 3, W[2],  bs[2], buf1, 1024, 1024, 1, 2);

    // concat [buf1(1024) | x1b(128)] -> buf0 (width 1152)
    long t1 = rows * 1024;
    copy_cols_kernel<<<(t1 + 255) / 256, 256, 0, stream>>>(
        buf1, buf0, t1, 1024, 1024, 1152, Nn,
        (long)Nn * 1024, (long)Nn * 1152);
    long t2 = rows * 128;
    copy_cols_kernel<<<(t2 + 255) / 256, 256, 0, stream>>>(
        x1b, buf0 + 1024, t2, 128, 128, 1152, Nn,
        (long)Nn * 128, (long)Nn * 1152);

    run_layer(p, buf0, 1152, 1, W[3],  bs[3], buf1, 512, 512, 1, 3);
    run_layer(p, buf1, 512,  1, W[4],  bs[4], buf0, 128, 128, 1, 4);
    run_layer(p, buf0, 128,  1, w6pad, bs[5], out,  64,  50,  0, 5);
}
